// GeometricNetwork_5815385719081
// MI455X (gfx1250) — compile-verified, matrix operand order checked by reading, not run on hardware
//
#include <hip/hip_runtime.h>

// ---------------------------------------------------------------------------
// GNN MetaLayer forward on gfx1250 (MI455X), wave32 + WMMA f16->f32.
// One wave-tiled WMMA GEMM with "segment" A operands (gather fused into the
// A-fragment load). v2: 2 M-tiles per wave (B-fragment reuse, 16 independent
// WMMAs per k-step) and segment select without a stack array so A pointers
// infer as global (global_load_b128 instead of flat_load_b128).
// ---------------------------------------------------------------------------

typedef __attribute__((ext_vector_type(16))) _Float16 v16h;
typedef __attribute__((ext_vector_type(8)))  _Float16 v8h;
typedef __attribute__((ext_vector_type(8)))  float    v8f;

static constexpr int GN_B  = 1024;            // graphs
static constexpr int GN_N  = 16;              // nodes / graph
static constexpr int GN_V  = GN_B * GN_N;     // 16384 nodes
static constexpr int GN_EPG= GN_N * (GN_N-1); // 240 edges / graph
static constexpr int GN_E  = GN_B * GN_EPG;   // 245760 edges
static constexpr int GN_H  = 128;
static constexpr int GN_NF = 12, GN_EF = 6, GN_XC = 7, GN_EC = 6;

struct GnSeg { const _Float16* a; const int* idx; int lda; int ksteps; };

__device__ __forceinline__ v16h gn_combine(v8h lo, v8h hi) {
  v16h r;
#pragma unroll
  for (int j = 0; j < 8; ++j) { r[j] = lo[j]; r[j + 8] = hi[j]; }
  return r;
}

// ---------------------------------------------------------------------------
// WMMA GEMM: C[M, NT*16] = concat_k(segments) x Wt^T + bias  (opt ReLU)
// Wt pre-packed [Npad, Kpad] f16 (W transposed + zero padded).
// Each wave computes TWO 16-row tiles across NT 16-col tiles.
// ---------------------------------------------------------------------------
template <int NT>
__global__ __launch_bounds__(256) void gn_gemm(
    GnSeg s0, GnSeg s1, GnSeg s2, GnSeg s3, int nseg,
    const _Float16* __restrict__ Wt, int Kpad,
    const float* __restrict__ bias,
    float* __restrict__ Cf32, _Float16* __restrict__ Cf16, int ldc,
    int Mtiles, int relu)
{
  const int wave = blockIdx.x * (blockDim.x >> 5) + (threadIdx.x >> 5);
  const int mt0  = wave * 2;                  // two 16-row tiles per wave
  if (mt0 >= Mtiles) return;                  // wave-uniform: EXEC stays full
  const int lane = threadIdx.x & 31;
  const int r    = lane & 15;
  const int half = lane >> 4;
  const int m0   = mt0 * 16;

  v8f acc0[NT], acc1[NT];
#pragma unroll
  for (int nt = 0; nt < NT; ++nt)
#pragma unroll
    for (int j = 0; j < 8; ++j) { acc0[nt][j] = 0.0f; acc1[nt][j] = 0.0f; }

  int kglob = 0;
#pragma unroll
  for (int s = 0; s < 4; ++s) {
    // select by value from kernarg SSA (no alloca -> pointers infer global)
    const GnSeg sg = (s == 0) ? s0 : (s == 1) ? s1 : (s == 2) ? s2 : s3;
    if (s < nseg) {
      const int row0 = sg.idx ? sg.idx[m0 + r]      : (m0 + r);
      const int row1 = sg.idx ? sg.idx[m0 + 16 + r] : (m0 + 16 + r);
      const _Float16* ar0 = sg.a + (size_t)row0 * sg.lda + half * 8;
      const _Float16* ar1 = sg.a + (size_t)row1 * sg.lda + half * 8;
      for (int kt = 0; kt < sg.ksteps; ++kt) {
        const _Float16* ap0 = ar0 + kt * 32;
        const _Float16* ap1 = ar1 + kt * 32;
        __builtin_prefetch(ap0 + 64, 0, 1);
        __builtin_prefetch(ap1 + 64, 0, 1);
        // A frag (16x32 f16): lanes 0-15 -> K 0..7|16..23 ; lanes 16-31 -> +8
        v16h a0 = gn_combine(*(const v8h*)(ap0), *(const v8h*)(ap0 + 16));
        v16h a1 = gn_combine(*(const v8h*)(ap1), *(const v8h*)(ap1 + 16));
        const _Float16* wp = Wt + (size_t)r * Kpad + kglob + kt * 32 + half * 8;
#pragma unroll
        for (int nt = 0; nt < NT; ++nt) {
          const _Float16* wq = wp + (size_t)nt * 16 * Kpad;
          v16h b = gn_combine(*(const v8h*)(wq), *(const v8h*)(wq + 16));
          acc0[nt] = __builtin_amdgcn_wmma_f32_16x16x32_f16(
              false, a0, false, b, (short)0, acc0[nt], false, false);
          acc1[nt] = __builtin_amdgcn_wmma_f32_16x16x32_f16(
              false, a1, false, b, (short)0, acc1[nt], false, false);
        }
      }
      kglob += sg.ksteps * 32;
    }
  }

  // C layout: lane (0-15)=col, VGPR rr -> rows rr (lanes 0-15) / rr+8 (16-31)
#pragma unroll
  for (int mt = 0; mt < 2; ++mt) {
    const int mb = m0 + mt * 16;
#pragma unroll
    for (int nt = 0; nt < NT; ++nt) {
      const int cn = nt * 16 + r;
      const float bv = bias ? bias[cn] : 0.0f;
#pragma unroll
      for (int rr = 0; rr < 8; ++rr) {
        const int rowo = mb + rr + half * 8;
        float v = (mt == 0 ? acc0[nt][rr] : acc1[nt][rr]) + bv;
        if (relu) v = fmaxf(v, 0.0f);
        if (Cf32) Cf32[(size_t)rowo * ldc + cn] = v;
        if (Cf16) Cf16[(size_t)rowo * ldc + cn] = (_Float16)v;
      }
    }
  }
}

// ------------------------------ utility kernels ----------------------------
__global__ void gn_zero16(_Float16* p, size_t n) {
  size_t t = (size_t)blockIdx.x * blockDim.x + threadIdx.x;
  if (t < n) p[t] = (_Float16)0.0f;
}
__global__ void gn_packw(const float* __restrict__ W, int dout,
                         _Float16* __restrict__ Wt, int Kpad,
                         int src0, int rows, int koff, size_t total) {
  size_t t = (size_t)blockIdx.x * blockDim.x + threadIdx.x;
  if (t >= total) return;
  int n  = (int)(t / rows);
  int kk = (int)(t % rows);
  Wt[(size_t)n * Kpad + koff + kk] = (_Float16)W[(size_t)(src0 + kk) * dout + n];
}
__global__ void gn_packb(const float* __restrict__ b, int dout,
                         float* __restrict__ bp, int Npad) {
  int n = blockIdx.x * blockDim.x + threadIdx.x;
  if (n < Npad) bp[n] = (n < dout) ? b[n] : 0.0f;
}
__global__ void gn_iota_div(int* p, int n, int d) {
  int t = blockIdx.x * blockDim.x + threadIdx.x;
  if (t < n) p[t] = t / d;
}
__global__ void gn_xin(const float* __restrict__ x, _Float16* __restrict__ o) {
  size_t t = (size_t)blockIdx.x * blockDim.x + threadIdx.x;
  if (t >= (size_t)GN_V * 32) return;
  int v = (int)(t >> 5), c = (int)(t & 31);
  o[t] = (c < GN_NF) ? (_Float16)x[(size_t)v * GN_NF + c] : (_Float16)0.0f;
}
__global__ void gn_upad(const float* __restrict__ u, _Float16* __restrict__ o) {
  size_t t = (size_t)blockIdx.x * blockDim.x + threadIdx.x;
  if (t >= (size_t)GN_B * 32) return;
  o[t] = ((t & 31) == 0) ? (_Float16)u[t >> 5] : (_Float16)0.0f;
}
__global__ void gn_concat_first_edge(const float* __restrict__ x,
                                     const float* __restrict__ ea,
                                     const float* __restrict__ u,
                                     const int* __restrict__ row,
                                     const int* __restrict__ col,
                                     _Float16* __restrict__ o) {
  size_t t = (size_t)blockIdx.x * blockDim.x + threadIdx.x;
  if (t >= (size_t)GN_E * 32) return;
  int e = (int)(t >> 5), c = (int)(t & 31);
  float v;
  if      (c < GN_NF)            v = x[(size_t)row[e] * GN_NF + c];
  else if (c < 2 * GN_NF)        v = x[(size_t)col[e] * GN_NF + (c - GN_NF)];
  else if (c < 2 * GN_NF + GN_EF)v = ea[(size_t)e * GN_EF + (c - 2 * GN_NF)];
  else if (c == 30)              v = u[e / GN_EPG];
  else                           v = 0.0f;
  o[t] = (_Float16)v;
}
// mean of incoming edge messages per node (exactly 15 per node)
__global__ void gn_mean_in(const float* __restrict__ e, int lde, int cols,
                           _Float16* __restrict__ agg, int ldagg, float inv) {
  size_t t = (size_t)blockIdx.x * blockDim.x + threadIdx.x;
  if (t >= (size_t)GN_V * cols) return;
  int v = (int)(t / cols), c = (int)(t % cols);
  int g = v >> 4, j = v & 15;
  float s = 0.0f;
  for (int i = 0; i < 16; ++i) {
    if (i == j) continue;
    int le = i * 15 + j - (j > i ? 1 : 0);
    s += e[(size_t)(g * GN_EPG + le) * lde + c];
  }
  agg[(size_t)v * ldagg + c] = (_Float16)(s * inv);
}
__global__ void gn_mean_nodes(const float* __restrict__ xs, int lds, int cols,
                              _Float16* __restrict__ g16, int ldg) {
  size_t t = (size_t)blockIdx.x * blockDim.x + threadIdx.x;
  if (t >= (size_t)GN_B * cols) return;
  int g = (int)(t / cols), c = (int)(t % cols);
  float s = 0.0f;
  for (int n = 0; n < 16; ++n) s += xs[(size_t)(g * 16 + n) * lds + c];
  g16[(size_t)g * ldg + c] = (_Float16)(s * (1.0f / 16.0f));
}
__global__ void gn_mean_edges(const float* __restrict__ es, int lds, int cols,
                              _Float16* __restrict__ g16, int ldg) {
  size_t t = (size_t)blockIdx.x * blockDim.x + threadIdx.x;
  if (t >= (size_t)GN_B * cols) return;
  int g = (int)(t / cols), c = (int)(t % cols);
  float s = 0.0f;
  for (int k = 0; k < GN_EPG; ++k) s += es[(size_t)(g * GN_EPG + k) * lds + c];
  g16[(size_t)g * ldg + c] = (_Float16)(s * (1.0f / (float)GN_EPG));
}
__global__ void gn_resid(float* __restrict__ d32, _Float16* __restrict__ d16,
                         const float* __restrict__ add, size_t n) {
  size_t t = (size_t)blockIdx.x * blockDim.x + threadIdx.x;
  if (t >= n) return;
  float v = d32[t] + add[t];
  d32[t] = v;
  d16[t] = (_Float16)v;
}
__global__ void gn_out_nodes(const float* __restrict__ xl, float* __restrict__ o) {
  size_t t = (size_t)blockIdx.x * blockDim.x + threadIdx.x;
  if (t >= (size_t)GN_V * GN_XC) return;
  int v = (int)(t / GN_XC), c = (int)(t % GN_XC);
  o[t] = xl[(size_t)v * 16 + c];
}
__global__ void gn_out_edges(const float* __restrict__ el,
                             const int* __restrict__ perm, float* __restrict__ o) {
  size_t t = (size_t)blockIdx.x * blockDim.x + threadIdx.x;
  if (t >= (size_t)GN_E * GN_EC) return;
  int e = (int)(t / GN_EC), c = (int)(t % GN_EC);
  o[t] = 0.5f * (el[(size_t)e * 16 + c] + el[(size_t)perm[e] * 16 + c]);
}
__global__ void gn_out_u(const float* __restrict__ ul, float* __restrict__ o) {
  int t = blockIdx.x * blockDim.x + threadIdx.x;
  if (t < GN_B) o[t] = ul[(size_t)t * 16];
}

// ------------------------------ host helpers -------------------------------
struct GnWSeg { int src0, rows, koff; };

static inline unsigned gn_nb(size_t n) { return (unsigned)((n + 255) / 256); }

static inline void gn_launch_gemm(hipStream_t st, int M, int Npad,
                                  const GnSeg* sg, int nseg,
                                  const _Float16* Wt, int Kpad, const float* bias,
                                  float* C32, _Float16* C16, int ldc, int relu) {
  GnSeg z = {nullptr, nullptr, 0, 0};
  GnSeg a0 = nseg > 0 ? sg[0] : z, a1 = nseg > 1 ? sg[1] : z;
  GnSeg a2 = nseg > 2 ? sg[2] : z, a3 = nseg > 3 ? sg[3] : z;
  int Mtiles = M / 16;                 // always even in this net
  int waves  = (Mtiles + 1) / 2;       // 2 M-tiles per wave
  int blocks = (waves + 7) / 8;        // 8 waves (256 threads) per block
  if (Npad == 128)
    gn_gemm<8><<<blocks, 256, 0, st>>>(a0,a1,a2,a3,nseg,Wt,Kpad,bias,C32,C16,ldc,Mtiles,relu);
  else
    gn_gemm<1><<<blocks, 256, 0, st>>>(a0,a1,a2,a3,nseg,Wt,Kpad,bias,C32,C16,ldc,Mtiles,relu);
}

static inline void gn_pack_linear(hipStream_t st, const float* W, const float* b,
                                  int dout, int Npad, int Kpad,
                                  const GnWSeg* segs, int nseg,
                                  _Float16* Wt, float* bp) {
  size_t tot = (size_t)Npad * Kpad;
  gn_zero16<<<gn_nb(tot), 256, 0, st>>>(Wt, tot);
  for (int s = 0; s < nseg; ++s) {
    size_t tn = (size_t)dout * segs[s].rows;
    gn_packw<<<gn_nb(tn), 256, 0, st>>>(W, dout, Wt, Kpad,
                                        segs[s].src0, segs[s].rows, segs[s].koff, tn);
  }
  gn_packb<<<gn_nb(Npad), 256, 0, st>>>(b, dout, bp, Npad);
}

// ---------------------------------------------------------------------------
extern "C" void kernel_launch(void* const* d_in, const int* in_sizes, int n_in,
                              void* d_out, int out_size, void* d_ws, size_t ws_size,
                              hipStream_t stream) {
  (void)in_sizes; (void)n_in; (void)out_size; (void)ws_size;
  const int B = GN_B, V = GN_V, E = GN_E, H = GN_H;

  const float* x_in  = (const float*)d_in[0];
  const float* ea_in = (const float*)d_in[1];
  const float* u_in  = (const float*)d_in[2];
  // params pytree leaves (dict keys sorted: first,last,mid; edge,glob,node;
  // each MLP: W0,b0,W1,b1,W2,b2)
  const int PFE = 3, PFG = 9, PFN = 15, PLE = 21, PLG = 27, PLN = 33,
            PME = 39, PMG = 45, PMN = 51;
  const int* erow  = (const int*)d_in[57];
  const int* ecol  = erow + E;
  const int* batch = (const int*)d_in[58];
  const int* perm  = (const int*)d_in[59];

  // ---- workspace allocator ----
  char* base = (char*)d_ws; size_t off = 0;
  auto alloc = [&](size_t bytes) -> void* {
    off = (off + 255) & ~(size_t)255;
    void* p = base + off; off += bytes; return p;
  };

  // ---- packed weights: MLP order FE FG FN LE LG LN ME MG MN, 3 layers each
  struct Lin { int pi; int dout; int Npad; int Kpad; int nseg; GnWSeg sg[3]; };
  const Lin L[27] = {
    // FE: 31 -> 128 -> 128 -> 128
    {PFE+0, H, H,  32, 1, {{0,31,0}}},
    {PFE+2, H, H,   H, 1, {{0,H,0}}},
    {PFE+4, H, H,   H, 1, {{0,H,0}}},
    // FG: 257 (xg128|eg128|u1->pad32)
    {PFG+0, H, H, 288, 3, {{0,H,0},{H,H,H},{256,1,256}}},
    {PFG+2, H, H,   H, 1, {{0,H,0}}},
    {PFG+4, H, H,   H, 1, {{0,H,0}}},
    // FN: 141 (x12->pad32|agg128|u1->pad32)
    {PFN+0, H, H, 192, 3, {{0,GN_NF,0},{GN_NF,H,32},{GN_NF+H,1,160}}},
    {PFN+2, H, H,   H, 1, {{0,H,0}}},
    {PFN+4, H, H,   H, 1, {{0,H,0}}},
    // LE: 512 -> 128 -> 128 -> 6(pad16)
    {PLE+0, H,     H, 512, 1, {{0,512,0}}},
    {PLE+2, H,     H,   H, 1, {{0,H,0}}},
    {PLE+4, GN_EC, 16,  H, 1, {{0,H,0}}},
    // LG: 141 (xg7->pad32|eg6->pad32|u128) -> ... -> 1(pad16)
    {PLG+0, H, H, 192, 3, {{0,GN_XC,0},{GN_XC,GN_EC,32},{GN_XC+GN_EC,H,64}}},
    {PLG+2, H, H,   H, 1, {{0,H,0}}},
    {PLG+4, 1, 16,  H, 1, {{0,H,0}}},
    // LN: 262 (x128|agg6->pad32|u128) -> ... -> 7(pad16)
    {PLN+0, H, H, 288, 3, {{0,H,0},{H,GN_EC,H},{H+GN_EC,H,160}}},
    {PLN+2, H, H,   H, 1, {{0,H,0}}},
    {PLN+4, GN_XC, 16, H, 1, {{0,H,0}}},
    // ME: 512
    {PME+0, H, H, 512, 1, {{0,512,0}}},
    {PME+2, H, H,   H, 1, {{0,H,0}}},
    {PME+4, H, H,   H, 1, {{0,H,0}}},
    // MG: 384
    {PMG+0, H, H, 384, 1, {{0,384,0}}},
    {PMG+2, H, H,   H, 1, {{0,H,0}}},
    {PMG+4, H, H,   H, 1, {{0,H,0}}},
    // MN: 384
    {PMN+0, H, H, 384, 1, {{0,384,0}}},
    {PMN+2, H, H,   H, 1, {{0,H,0}}},
    {PMN+4, H, H,   H, 1, {{0,H,0}}},
  };
  _Float16* WT[27]; float* BP[27];
  for (int i = 0; i < 27; ++i) {
    WT[i] = (_Float16*)alloc((size_t)L[i].Npad * L[i].Kpad * sizeof(_Float16));
    BP[i] = (float*)   alloc((size_t)L[i].Npad * sizeof(float));
    gn_pack_linear(stream, (const float*)d_in[L[i].pi], (const float*)d_in[L[i].pi + 1],
                   L[i].dout, L[i].Npad, L[i].Kpad, L[i].sg, L[i].nseg, WT[i], BP[i]);
  }

  // ---- activation buffers ----
  _Float16* X16  = (_Float16*)alloc((size_t)V * H * 2);
  float*    XF32 = (float*)   alloc((size_t)V * H * 4);
  _Float16* E16  = (_Float16*)alloc((size_t)E * H * 2);
  float*    EF32 = (float*)   alloc((size_t)E * H * 4);
  _Float16* U16  = (_Float16*)alloc((size_t)B * H * 2);
  float*    UF32 = (float*)   alloc((size_t)B * H * 4);
  _Float16* H0   = (_Float16*)alloc((size_t)E * H * 2);
  _Float16* H1   = (_Float16*)alloc((size_t)E * H * 2);
  float*    T0   = (float*)   alloc((size_t)E * H * 4);  // en_f32 / last-stage outs
  float*    XN32 = (float*)   alloc((size_t)V * H * 4);
  float*    UN32 = (float*)   alloc((size_t)B * H * 4);
  _Float16* AGG  = (_Float16*)alloc((size_t)V * H * 2);
  _Float16* XG   = (_Float16*)alloc((size_t)B * H * 2);
  _Float16* EG   = (_Float16*)alloc((size_t)B * H * 2);
  _Float16* UPAD = (_Float16*)alloc((size_t)B * 32 * 2);
  _Float16* XIN  = (_Float16*)alloc((size_t)V * 32 * 2);
  _Float16* AGGL = (_Float16*)alloc((size_t)V * 32 * 2);
  _Float16* XGL  = (_Float16*)alloc((size_t)B * 32 * 2);
  _Float16* EGL  = (_Float16*)alloc((size_t)B * 32 * 2);
  int*      EB   = (int*)     alloc((size_t)E * 4);

  // ---- input prep ----
  gn_iota_div<<<gn_nb(E), 256, 0, stream>>>(EB, E, GN_EPG);
  gn_xin <<<gn_nb((size_t)V * 32), 256, 0, stream>>>(x_in, XIN);
  gn_upad<<<gn_nb((size_t)B * 32), 256, 0, stream>>>(u_in, UPAD);
  _Float16* EIN = (_Float16*)T0;   // E x 32 f16, reuses T0 region
  gn_concat_first_edge<<<gn_nb((size_t)E * 32), 256, 0, stream>>>(
      x_in, ea_in, u_in, erow, ecol, EIN);

  GnSeg s[4];
  // ================= FIRST meta layer =================
  s[0] = {EIN, nullptr, 32, 1};
  gn_launch_gemm(stream, E, H, s, 1, WT[0], 32, BP[0], nullptr, H0, H, 1);
  s[0] = {H0, nullptr, H, 4};
  gn_launch_gemm(stream, E, H, s, 1, WT[1], H, BP[1], nullptr, H1, H, 1);
  s[0] = {H1, nullptr, H, 4};
  gn_launch_gemm(stream, E, H, s, 1, WT[2], H, BP[2], EF32, E16, H, 0);
  gn_mean_in<<<gn_nb((size_t)V * H), 256, 0, stream>>>(EF32, H, H, AGG, H, 1.0f/15.0f);
  s[0] = {XIN, nullptr, 32, 1}; s[1] = {AGG, nullptr, H, 4}; s[2] = {UPAD, batch, 32, 1};
  gn_launch_gemm(stream, V, H, s, 3, WT[6], 192, BP[6], nullptr, H0, H, 1);
  s[0] = {H0, nullptr, H, 4};
  gn_launch_gemm(stream, V, H, s, 1, WT[7], H, BP[7], nullptr, H1, H, 1);
  s[0] = {H1, nullptr, H, 4};
  gn_launch_gemm(stream, V, H, s, 1, WT[8], H, BP[8], XF32, X16, H, 0);
  gn_mean_nodes<<<gn_nb((size_t)B * H), 256, 0, stream>>>(XF32, H, H, XG, H);
  gn_mean_edges<<<gn_nb((size_t)B * H), 256, 0, stream>>>(EF32, H, H, EG, H);
  s[0] = {XG, nullptr, H, 4}; s[1] = {EG, nullptr, H, 4}; s[2] = {UPAD, nullptr, 32, 1};
  gn_launch_gemm(stream, B, H, s, 3, WT[3], 288, BP[3], nullptr, H0, H, 1);
  s[0] = {H0, nullptr, H, 4};
  gn_launch_gemm(stream, B, H, s, 1, WT[4], H, BP[4], nullptr, H1, H, 1);
  s[0] = {H1, nullptr, H, 4};
  gn_launch_gemm(stream, B, H, s, 1, WT[5], H, BP[5], UF32, U16, H, 0);

  // ================= MID meta layer (residual) =================
  s[0] = {X16, erow, H, 4}; s[1] = {X16, ecol, H, 4};
  s[2] = {E16, nullptr, H, 4}; s[3] = {U16, EB, H, 4};
  gn_launch_gemm(stream, E, H, s, 4, WT[18], 512, BP[18], nullptr, H0, H, 1);
  s[0] = {H0, nullptr, H, 4};
  gn_launch_gemm(stream, E, H, s, 1, WT[19], H, BP[19], nullptr, H1, H, 1);
  s[0] = {H1, nullptr, H, 4};
  gn_launch_gemm(stream, E, H, s, 1, WT[20], H, BP[20], T0, nullptr, H, 0); // en
  gn_mean_in<<<gn_nb((size_t)V * H), 256, 0, stream>>>(T0, H, H, AGG, H, 1.0f/15.0f);
  s[0] = {X16, nullptr, H, 4}; s[1] = {AGG, nullptr, H, 4}; s[2] = {U16, batch, H, 4};
  gn_launch_gemm(stream, V, H, s, 3, WT[24], 384, BP[24], nullptr, H0, H, 1);
  s[0] = {H0, nullptr, H, 4};
  gn_launch_gemm(stream, V, H, s, 1, WT[25], H, BP[25], nullptr, H1, H, 1);
  s[0] = {H1, nullptr, H, 4};
  gn_launch_gemm(stream, V, H, s, 1, WT[26], H, BP[26], XN32, nullptr, H, 0); // xn
  gn_mean_nodes<<<gn_nb((size_t)B * H), 256, 0, stream>>>(XN32, H, H, XG, H);
  gn_mean_edges<<<gn_nb((size_t)B * H), 256, 0, stream>>>(T0, H, H, EG, H);
  s[0] = {XG, nullptr, H, 4}; s[1] = {EG, nullptr, H, 4}; s[2] = {U16, nullptr, H, 4};
  gn_launch_gemm(stream, B, H, s, 3, WT[21], 384, BP[21], nullptr, H0, H, 1);
  s[0] = {H0, nullptr, H, 4};
  gn_launch_gemm(stream, B, H, s, 1, WT[22], H, BP[22], nullptr, H1, H, 1);
  s[0] = {H1, nullptr, H, 4};
  gn_launch_gemm(stream, B, H, s, 1, WT[23], H, BP[23], UN32, nullptr, H, 0); // un
  gn_resid<<<gn_nb((size_t)E * H), 256, 0, stream>>>(EF32, E16, T0,   (size_t)E * H);
  gn_resid<<<gn_nb((size_t)V * H), 256, 0, stream>>>(XF32, X16, XN32, (size_t)V * H);
  gn_resid<<<gn_nb((size_t)B * H), 256, 0, stream>>>(UF32, U16, UN32, (size_t)B * H);

  // ================= LAST meta layer =================
  float* ELAST = T0;                                // [E,16] f32 (6 valid)
  float* XLAST = T0 + (size_t)E * 16;               // [V,16] f32 (7 valid)
  float* ULAST = XLAST + (size_t)V * 16;            // [B,16] f32 (1 valid)
  s[0] = {X16, erow, H, 4}; s[1] = {X16, ecol, H, 4};
  s[2] = {E16, nullptr, H, 4}; s[3] = {U16, EB, H, 4};
  gn_launch_gemm(stream, E, H, s, 4, WT[9], 512, BP[9], nullptr, H0, H, 1);
  s[0] = {H0, nullptr, H, 4};
  gn_launch_gemm(stream, E, H, s, 1, WT[10], H, BP[10], nullptr, H1, H, 1);
  s[0] = {H1, nullptr, H, 4};
  gn_launch_gemm(stream, E, 16, s, 1, WT[11], H, BP[11], ELAST, nullptr, 16, 0);
  gn_zero16<<<gn_nb((size_t)V * 32), 256, 0, stream>>>(AGGL, (size_t)V * 32);
  gn_mean_in<<<gn_nb((size_t)V * GN_EC), 256, 0, stream>>>(ELAST, 16, GN_EC, AGGL, 32, 1.0f/15.0f);
  s[0] = {X16, nullptr, H, 4}; s[1] = {AGGL, nullptr, 32, 1}; s[2] = {U16, batch, H, 4};
  gn_launch_gemm(stream, V, H, s, 3, WT[15], 288, BP[15], nullptr, H0, H, 1);
  s[0] = {H0, nullptr, H, 4};
  gn_launch_gemm(stream, V, H, s, 1, WT[16], H, BP[16], nullptr, H1, H, 1);
  s[0] = {H1, nullptr, H, 4};
  gn_launch_gemm(stream, V, 16, s, 1, WT[17], H, BP[17], XLAST, nullptr, 16, 0);
  gn_zero16<<<gn_nb((size_t)B * 32), 256, 0, stream>>>(XGL, (size_t)B * 32);
  gn_zero16<<<gn_nb((size_t)B * 32), 256, 0, stream>>>(EGL, (size_t)B * 32);
  gn_mean_nodes<<<gn_nb((size_t)B * GN_XC), 256, 0, stream>>>(XLAST, 16, GN_XC, XGL, 32);
  gn_mean_edges<<<gn_nb((size_t)B * GN_EC), 256, 0, stream>>>(ELAST, 16, GN_EC, EGL, 32);
  s[0] = {XGL, nullptr, 32, 1}; s[1] = {EGL, nullptr, 32, 1}; s[2] = {U16, nullptr, H, 4};
  gn_launch_gemm(stream, B, H, s, 3, WT[12], 192, BP[12], nullptr, H0, H, 1);
  s[0] = {H0, nullptr, H, 4};
  gn_launch_gemm(stream, B, H, s, 1, WT[13], H, BP[13], nullptr, H1, H, 1);
  s[0] = {H1, nullptr, H, 4};
  gn_launch_gemm(stream, B, 16, s, 1, WT[14], H, BP[14], ULAST, nullptr, 16, 0);

  // ---- outputs: x [V,7] | symmetrized edges [E,6] | u [B,1] ----
  float* out = (float*)d_out;
  gn_out_nodes<<<gn_nb((size_t)V * GN_XC), 256, 0, stream>>>(XLAST, out);
  gn_out_edges<<<gn_nb((size_t)E * GN_EC), 256, 0, stream>>>(
      ELAST, perm, out + (size_t)V * GN_XC);
  gn_out_u<<<gn_nb(GN_B), 256, 0, stream>>>(
      ULAST, out + (size_t)V * GN_XC + (size_t)E * GN_EC);
}
